// GCN_57853209477558
// MI455X (gfx1250) — matrix operand order repaired and checked by reading, not compiled, not run on hardware
//
#include <hip/hip_runtime.h>

#define NN   16384
#define DIN  128
#define DHID 128
#define DOUT 64

typedef __attribute__((ext_vector_type(16))) __bf16 v16bf;
typedef __attribute__((ext_vector_type(8)))  float  v8f;
typedef __attribute__((ext_vector_type(4)))  float  v4f;

// ---------------------------------------------------------------------------
// Kernel 1: Pt1[c][n] = bf16( dot(X[n,:], W1[:,c]) )   (transposed bf16 store)
// ---------------------------------------------------------------------------
__global__ void __launch_bounds__(128) k_xw1(const float* __restrict__ X,
                                             const float* __restrict__ W1,
                                             __bf16* __restrict__ Pt1)
{
    __shared__ float xs[DIN];
    const int n = blockIdx.x;
    const int c = threadIdx.x;
    xs[c] = X[(size_t)n * DIN + c];
    __syncthreads();
    float s = 0.f;
#pragma unroll 8
    for (int k = 0; k < DIN; ++k) s += xs[k] * W1[k * DHID + c];
    Pt1[(size_t)c * NN + n] = (__bf16)s;
}

// ---------------------------------------------------------------------------
// Software-pipelined WMMA aggregation stages.
//
// A-matrix layout chosen here (16-bit A 16x32, contiguous k order, same as B):
//   lanes 0-15 : row = m0+lane,    elems 0..15 = K k..k+15
//   lanes16-31 : row = m0+lane-16, elems 0..15 = K k+16..k+31
// B-matrix: lane (mod 16) = column, K contiguous per lane; lanes>=16 K+16..K+31.
// ---------------------------------------------------------------------------
template <int NCT>
__device__ __forceinline__ void stage_load(const float*  __restrict__ arow,
                                           const __bf16* __restrict__ bbase,
                                           int k, int lo, v4f a[4], v16bf b[NCT])
{
    // A_hat stream first (HBM, single-use -> non-temporal); completes first
    // in-order so the bf16 conversion can start while B is still in flight.
    a[0] = __builtin_nontemporal_load((const v4f*)(arow + k));
    a[1] = __builtin_nontemporal_load((const v4f*)(arow + k + 4));
    a[2] = __builtin_nontemporal_load((const v4f*)(arow + k + 8));
    a[3] = __builtin_nontemporal_load((const v4f*)(arow + k + 12));
    // All B tiles for this k-step issued together (L2-resident Pt)
#pragma unroll
    for (int t = 0; t < NCT; ++t)
        b[t] = *(const v16bf*)(bbase + (size_t)(t * 16 + lo) * NN + k);
}

template <int NCT>
__device__ __forceinline__ void stage_compute(const v4f a[4], const v16bf b[NCT],
                                              v8f acc[NCT])
{
    v16bf av;
#pragma unroll
    for (int i = 0; i < 4; ++i) {
        av[i]      = (__bf16)a[0][i];
        av[4 + i]  = (__bf16)a[1][i];
        av[8 + i]  = (__bf16)a[2][i];
        av[12 + i] = (__bf16)a[3][i];
    }
#pragma unroll
    for (int t = 0; t < NCT; ++t)
        acc[t] = __builtin_amdgcn_wmma_f32_16x16x32_bf16(
                     false, av, false, b[t], (short)0, acc[t], false, false);
}

// ---------------------------------------------------------------------------
// Aggregation: acc_out += A_hat[mtile rows, kseg] @ P  (P given transposed bf16)
// One wave: 16-row tile x NCT 16-col tiles; K split KSEG ways, fp32 atomics.
// Stage-level double buffering: the next stage's loads (4 NT A + 2*NCT B) are
// issued before the current stage's WMMAs, so waits become incremental.
// amdgpu_waves_per_eu(1) lifts the VGPR cap so BOTH stages stay live
// (~230 VGPRs) instead of the allocator recycling one staging buffer.
// ---------------------------------------------------------------------------
template <int NCT, int KSEG>
__global__ void __launch_bounds__(256)
__attribute__((amdgpu_waves_per_eu(1)))
k_agg(const float*  __restrict__ A,
      const __bf16* __restrict__ Pt,
      float*        __restrict__ acc_out)
{
    const int wave  = (int)((blockIdx.x * blockDim.x + threadIdx.x) >> 5);
    const int lane  = threadIdx.x & 31;
    const int mtile = wave % (NN / 16);
    const int kseg  = wave / (NN / 16);
    const int klen  = NN / KSEG;     // multiple of 64
    const int k0    = kseg * klen;

    const int hi = lane >> 4;        // 0 or 1
    const int lo = lane & 15;

    const float*  __restrict__ arow  = A  + (size_t)(mtile * 16 + lo) * NN + k0 + hi * 16;
    const __bf16* __restrict__ bbase = Pt + (size_t)k0 + hi * 16;

    v8f acc[NCT];
    const v8f vzero = {0.f, 0.f, 0.f, 0.f, 0.f, 0.f, 0.f, 0.f};
#pragma unroll
    for (int t = 0; t < NCT; ++t) acc[t] = vzero;

    v4f   aA[4], aB[4];
    v16bf bA[NCT], bB[NCT];

    stage_load<NCT>(arow, bbase, 0, lo, aA, bA);
    int k = 0;
    for (; k < klen - 64; k += 64) {
        stage_load<NCT>(arow, bbase, k + 32, lo, aB, bB);
        stage_compute<NCT>(aA, bA, acc);
        stage_load<NCT>(arow, bbase, k + 64, lo, aA, bA);
        stage_compute<NCT>(aB, bB, acc);
    }
    // peeled final iteration (no out-of-segment prefetch)
    stage_load<NCT>(arow, bbase, k + 32, lo, aB, bB);
    stage_compute<NCT>(aA, bA, acc);
    stage_compute<NCT>(aB, bB, acc);

    // C/D layout: VGPR i, lanes0-15 -> M=i, lanes16-31 -> M=i+8; N = lane&15
    const int orow = mtile * 16 + hi * 8;
#pragma unroll
    for (int t = 0; t < NCT; ++t) {
        const int ocol = t * 16 + lo;
#pragma unroll
        for (int i = 0; i < 8; ++i)
            atomicAdd(&acc_out[(size_t)(orow + i) * (NCT * 16) + ocol], acc[t][i]);
    }
}

// ---------------------------------------------------------------------------
// Kernel 3: H[n][c] = relu(acc1[n][c] + b1[c]);  Pt2[o][n] = bf16(H[n,:] @ W2[:,o])
// ---------------------------------------------------------------------------
__global__ void __launch_bounds__(128) k_relu_w2(const float* __restrict__ acc1,
                                                 const float* __restrict__ b1,
                                                 const float* __restrict__ W2,
                                                 __bf16* __restrict__ Pt2)
{
    __shared__ float hs[DHID];
    const int n = blockIdx.x;
    const int c = threadIdx.x;
    float h = acc1[(size_t)n * DHID + c] + b1[c];
    hs[c] = h > 0.f ? h : 0.f;
    __syncthreads();
    if (c < DOUT) {
        float s = 0.f;
#pragma unroll 8
        for (int k = 0; k < DHID; ++k) s += hs[k] * W2[k * DOUT + c];
        Pt2[(size_t)c * NN + n] = (__bf16)s;
    }
}

// ---------------------------------------------------------------------------
// Kernel 4: out[n][o] = b2[o]  (then k_agg<4,8> atomically adds A@P2)
// ---------------------------------------------------------------------------
__global__ void __launch_bounds__(256) k_init_out(const float* __restrict__ b2,
                                                  float* __restrict__ out)
{
    const size_t i = (size_t)blockIdx.x * blockDim.x + threadIdx.x;
    if (i < (size_t)NN * DOUT) out[i] = b2[i & (DOUT - 1)];
}

// ---------------------------------------------------------------------------
extern "C" void kernel_launch(void* const* d_in, const int* in_sizes, int n_in,
                              void* d_out, int out_size, void* d_ws, size_t ws_size,
                              hipStream_t stream)
{
    const float* X  = (const float*)d_in[0];
    const float* A  = (const float*)d_in[1];
    const float* W1 = (const float*)d_in[2];
    const float* b1 = (const float*)d_in[3];
    const float* W2 = (const float*)d_in[4];
    const float* b2 = (const float*)d_in[5];
    float* out = (float*)d_out;

    char* ws = (char*)d_ws;
    float*  acc1 = (float*)ws;                                    // 16384*128*4 = 8 MB
    __bf16* Pt1  = (__bf16*)(ws + (size_t)8  * 1024 * 1024);      // 128*16384*2 = 4 MB
    __bf16* Pt2  = (__bf16*)(ws + (size_t)12 * 1024 * 1024);      // 64*16384*2  = 2 MB

    // P1^T = bf16(X @ W1)
    k_xw1<<<NN, 128, 0, stream>>>(X, W1, Pt1);

    // acc1 = 0, then acc1 += A @ P1   (8 col tiles, K split 4 -> 4096 waves)
    hipMemsetAsync(acc1, 0, (size_t)NN * DHID * sizeof(float), stream);
    {
        const int waves = (NN / 16) * 4;
        k_agg<8, 4><<<waves * 32 / 256, 256, 0, stream>>>(A, Pt1, acc1);
    }

    // H = relu(acc1 + b1);  P2^T = bf16(H @ W2)
    k_relu_w2<<<NN, 128, 0, stream>>>(acc1, b1, W2, Pt2);

    // out = b2, then out += A @ P2    (4 col tiles, K split 8 -> 8192 waves)
    k_init_out<<<(NN * DOUT) / 256, 256, 0, stream>>>(b2, out);
    {
        const int waves = (NN / 16) * 8;
        k_agg<4, 8><<<waves * 32 / 256, 256, 0, stream>>>(A, Pt2, out);
    }
}
